// Net_70720931496759
// MI455X (gfx1250) — matrix operand order repaired and checked by reading, not compiled, not run on hardware
//
#include <hip/hip_runtime.h>
#include <hip/hip_bf16.h>

typedef __attribute__((ext_vector_type(16))) _Float16 v16h;
typedef __attribute__((ext_vector_type(8)))  _Float16 v8h;
typedef __attribute__((ext_vector_type(8)))  float    v8f;

#define GEMM_WAVES 4   // waves per block; each wave owns a 16x64 output strip

// ---------------------------------------------------------------------------
// Degree / normalization
// ---------------------------------------------------------------------------
__global__ void deg_init(float* __restrict__ deg, int N) {
    int i = blockIdx.x * blockDim.x + threadIdx.x;
    if (i < N) deg[i] = 1.0f;                       // self-loop contributes 1
}

__global__ void deg_count(float* __restrict__ deg, const int* __restrict__ cols, int E) {
    int e = blockIdx.x * blockDim.x + threadIdx.x;
    if (e < E) atomicAdd(&deg[cols[e]], 1.0f);
}

__global__ void deg_finalize(float* __restrict__ dinv, int N) {
    int i = blockIdx.x * blockDim.x + threadIdx.x;
    if (i < N) {
        float d = dinv[i];
        dinv[i] = (d > 0.0f) ? rsqrtf(d) : 0.0f;
    }
}

// ---------------------------------------------------------------------------
// Operand staging: f32 -> f16 with zero padding
// ---------------------------------------------------------------------------
__global__ void pad_x_f16(const float* __restrict__ x, _Float16* __restrict__ xh,
                          int N, int K, int KPAD)
{
    int idx = blockIdx.x * blockDim.x + threadIdx.x;
    int node = idx / KPAD, k = idx % KPAD;
    if (node >= N) return;
    xh[idx] = (_Float16)((k < K) ? x[(size_t)node * K + k] : 0.0f);
}

// Wt[FPAD][KPAD] = transpose(W[K][Fw]) with zero padding
__global__ void transpose_w_f16(const float* __restrict__ W, _Float16* __restrict__ Wt,
                                int K, int Fw, int KPAD, int FPAD)
{
    int idx = blockIdx.x * blockDim.x + threadIdx.x;
    int n = idx / KPAD, k = idx % KPAD;
    if (n >= FPAD) return;
    Wt[idx] = (_Float16)((n < Fw && k < K) ? W[(size_t)k * Fw + n] : 0.0f);
}

// ---------------------------------------------------------------------------
// GEMM: H[N x LDH] = Xh[N x KPAD] * Wt^T   (f16 WMMA, f32 accumulate)
// One wave -> 16x64 strip (4 adjacent 16x16 N-tiles), A fragment reused 4x.
// Branch-free 128-bit fragment loads; LDH templated so the store epilogue
// uses one base address per tile column + immediate offsets.
// ---------------------------------------------------------------------------
template<int KPAD, int LDH>
__global__ void __launch_bounds__(32 * GEMM_WAVES)
gemm_wmma_f16(const _Float16* __restrict__ Xh, const _Float16* __restrict__ Wt,
              float* __restrict__ H, int Nrows, int mtiles, int ntiles4)
{
    const int lane = threadIdx.x & 31;
    const int wid  = threadIdx.x >> 5;
    const int tile = blockIdx.x * GEMM_WAVES + wid;
    if (tile >= mtiles * ntiles4) return;            // uniform per wave

    const int mt   = tile / ntiles4;
    const int nt4  = tile % ntiles4;
    const int half = lane >> 4;                      // 0: lanes 0-15, 1: lanes 16-31
    const int l16  = lane & 15;

    int m = mt * 16 + l16;
    if (m >= Nrows) m = Nrows - 1;                   // clamp (tail tile only)
    const _Float16* xrow = Xh + (unsigned)m * KPAD;

    const _Float16* wrow[4];
    #pragma unroll
    for (int j = 0; j < 4; ++j)
        wrow[j] = Wt + (unsigned)(nt4 * 64 + j * 16 + l16) * KPAD;

    v8f acc[4] = {};

    #pragma unroll
    for (int k0 = 0; k0 < KPAD; k0 += 32) {
        // ---- A fragment: two contiguous 16B runs ----
        v8h a0 = *(const v8h*)(xrow + k0 + half * 8);
        v8h a1 = *(const v8h*)(xrow + k0 + 16 + half * 8);
        v16h a;
        #pragma unroll
        for (int e = 0; e < 8; ++e) { a[e] = a0[e]; a[e + 8] = a1[e]; }

        // ---- 4 B fragments, 1 WMMA each (A reused) ----
        #pragma unroll
        for (int j = 0; j < 4; ++j) {
            v8h b0 = *(const v8h*)(wrow[j] + k0 + half * 8);
            v8h b1 = *(const v8h*)(wrow[j] + k0 + 16 + half * 8);
            v16h b;
            #pragma unroll
            for (int e = 0; e < 8; ++e) { b[e] = b0[e]; b[e + 8] = b1[e]; }
            acc[j] = __builtin_amdgcn_wmma_f32_16x16x32_f16(
                false, a, false, b, (short)0, acc[j], false, false);
        }
    }

    // ---- store: C/D layout M=(lane>>4)*8+r, N=lane&15 ----
    const int row0 = mt * 16 + half * 8;
    const bool full = (mt * 16 + 16) <= Nrows;       // uniform per wave
    #pragma unroll
    for (int j = 0; j < 4; ++j) {
        float* hp = H + (unsigned)row0 * LDH + (unsigned)(nt4 * 64 + j * 16 + l16);
        if (full) {
            #pragma unroll
            for (int r = 0; r < 8; ++r) hp[r * LDH] = acc[j][r];
        } else {
            #pragma unroll
            for (int r = 0; r < 8; ++r)
                if (row0 + r < Nrows) hp[r * LDH] = acc[j][r];
        }
    }
}

// ---------------------------------------------------------------------------
// agg[i,:] = dinv[i]^2 * H[i,:]   (self-loop term; also initializes agg)
// ---------------------------------------------------------------------------
__global__ void agg_init(const float* __restrict__ H, const float* __restrict__ dinv,
                         float* __restrict__ agg, int N, int Fq)
{
    int idx  = blockIdx.x * blockDim.x + threadIdx.x;
    int node = idx / Fq;
    if (node >= N) return;
    float d = dinv[node];
    float w = d * d;
    const float4 v = ((const float4*)H)[idx];
    float4 o;
    o.x = w * v.x; o.y = w * v.y; o.z = w * v.z; o.w = w * v.w;
    ((float4*)agg)[idx] = o;
}

// ---------------------------------------------------------------------------
// Edge scatter: one wave per edge. agg[c,:] += dinv[r]*dinv[c] * H[r,:]
// float4 gathers; scalar f32 global atomics (resolve in 192MB L2).
// ---------------------------------------------------------------------------
__global__ void edge_scatter(const int* __restrict__ rows, const int* __restrict__ cols,
                             const float* __restrict__ dinv, const float* __restrict__ H,
                             float* __restrict__ agg, int E, int Fq, int ldh)
{
    int gtid = blockIdx.x * blockDim.x + threadIdx.x;
    int e    = gtid >> 5;
    int lane = gtid & 31;
    if (e >= E) return;                              // uniform per wave
    int r = rows[e];
    int c = cols[e];
    float w = dinv[r] * dinv[c];
    const float4* src = (const float4*)(H + (size_t)r * ldh);
    float*        dst = agg + (size_t)c * ldh;
    for (int f4 = lane; f4 < Fq; f4 += 32) {
        float4 v = src[f4];
        int f = f4 * 4;
        atomicAdd(dst + f + 0, w * v.x);
        atomicAdd(dst + f + 1, w * v.y);
        atomicAdd(dst + f + 2, w * v.z);
        atomicAdd(dst + f + 3, w * v.w);
    }
}

// ---------------------------------------------------------------------------
// bias + ReLU, emitting f16 activations for the next WMMA GEMM (ld == F)
// ---------------------------------------------------------------------------
__global__ void bias_relu_f16(const float* __restrict__ agg, const float* __restrict__ b,
                              _Float16* __restrict__ out, int N, int F)
{
    int idx = blockIdx.x * blockDim.x + threadIdx.x;
    if (idx >= N * F) return;
    int f = idx % F;
    float v = agg[idx] + b[f];
    out[idx] = (_Float16)fmaxf(v, 0.0f);
}

// final layer: bias only, f32, compact ldin -> F
__global__ void bias_out(const float* __restrict__ agg, const float* __restrict__ b,
                         float* __restrict__ out, int N, int F, int ldin)
{
    long long idx = (long long)blockIdx.x * blockDim.x + threadIdx.x;
    if (idx >= (long long)N * F) return;
    int node = (int)(idx / F);
    int f    = (int)(idx % F);
    out[idx] = agg[(size_t)node * ldin + f] + b[f];
}

// ---------------------------------------------------------------------------
extern "C" void kernel_launch(void* const* d_in, const int* in_sizes, int n_in,
                              void* d_out, int out_size, void* d_ws, size_t ws_size,
                              hipStream_t stream)
{
    const float* x  = (const float*)d_in[0];
    const int*   ei = (const int*)d_in[1];
    const float* W1 = (const float*)d_in[2];
    const float* b1 = (const float*)d_in[3];
    const float* W2 = (const float*)d_in[4];
    const float* b2 = (const float*)d_in[5];
    const float* W3 = (const float*)d_in[6];
    const float* b3 = (const float*)d_in[7];
    float* out = (float*)d_out;

    const int IN_F = 50, K1PAD = 64, HID = 256, OUT_F = 121, OUT_PAD = 128;
    const int N = in_sizes[0] / IN_F;                // 100000
    const int E = in_sizes[1] / 2;                   // 1600000
    const int* rows = ei;                            // edge_index[0] = source
    const int* cols = ei + E;                        // edge_index[1] = target

    // workspace layout (float units):
    // dinv | Hf32 (N*256) | Agg (N*256) | Xh f16 (N*256 halves) | Wt1 | Wt2 | Wt3
    float* ws = (float*)d_ws;
    size_t off = 0;
    float*    dinv = ws;                    off += ((size_t)N + 1023) & ~((size_t)1023);
    float*    Hf   = ws + off;              off += (size_t)N * HID;
    float*    Agg  = ws + off;              off += (size_t)N * HID;
    _Float16* Xh   = (_Float16*)(ws + off); off += (size_t)N * HID / 2;
    _Float16* Wt1  = (_Float16*)(ws + off); off += (size_t)HID * K1PAD / 2;     // [256][64]
    _Float16* Wt2  = (_Float16*)(ws + off); off += (size_t)HID * HID / 2;       // [256][256]
    _Float16* Wt3  = (_Float16*)(ws + off); off += (size_t)OUT_PAD * HID / 2;   // [128][256]

    const int TB = 256;
    const int mtiles = (N + 15) / 16;

    // --- normalization ---
    deg_init    <<<(N + TB - 1) / TB, TB, 0, stream>>>(dinv, N);
    deg_count   <<<(E + TB - 1) / TB, TB, 0, stream>>>(dinv, cols, E);
    deg_finalize<<<(N + TB - 1) / TB, TB, 0, stream>>>(dinv, N);

    // --- operand staging ---
    pad_x_f16<<<((N * K1PAD) + TB - 1) / TB, TB, 0, stream>>>(x, Xh, N, IN_F, K1PAD);
    transpose_w_f16<<<(HID * K1PAD + TB - 1) / TB, TB, 0, stream>>>(W1, Wt1, IN_F, HID, K1PAD, HID);
    transpose_w_f16<<<(HID * HID + TB - 1) / TB, TB, 0, stream>>>(W2, Wt2, HID, HID, HID, HID);
    transpose_w_f16<<<(OUT_PAD * HID + TB - 1) / TB, TB, 0, stream>>>(W3, Wt3, HID, OUT_F, HID, OUT_PAD);

    // ======================= layer 1 (K=50->64, F=256) ====================
    {
        int nt4   = (HID / 16) / 4;                                   // 4 strips
        int tiles = mtiles * nt4;
        gemm_wmma_f16<K1PAD, HID><<<(tiles + GEMM_WAVES - 1) / GEMM_WAVES, 32 * GEMM_WAVES, 0, stream>>>(
            Xh, Wt1, Hf, N, mtiles, nt4);
        int Fq = HID / 4;
        agg_init<<<(N * Fq + TB - 1) / TB, TB, 0, stream>>>(Hf, dinv, Agg, N, Fq);
        edge_scatter<<<(E * 32 + TB - 1) / TB, TB, 0, stream>>>(rows, cols, dinv, Hf, Agg, E, Fq, HID);
        bias_relu_f16<<<(N * HID + TB - 1) / TB, TB, 0, stream>>>(Agg, b1, Xh, N, HID);
    }

    // ======================= layer 2 (256 -> 256) =========================
    {
        int nt4   = (HID / 16) / 4;
        int tiles = mtiles * nt4;
        gemm_wmma_f16<HID, HID><<<(tiles + GEMM_WAVES - 1) / GEMM_WAVES, 32 * GEMM_WAVES, 0, stream>>>(
            Xh, Wt2, Hf, N, mtiles, nt4);
        int Fq = HID / 4;
        agg_init<<<(N * Fq + TB - 1) / TB, TB, 0, stream>>>(Hf, dinv, Agg, N, Fq);
        edge_scatter<<<(E * 32 + TB - 1) / TB, TB, 0, stream>>>(rows, cols, dinv, Hf, Agg, E, Fq, HID);
        bias_relu_f16<<<(N * HID + TB - 1) / TB, TB, 0, stream>>>(Agg, b2, Xh, N, HID);
    }

    // ======================= layer 3 (256 -> 121 pad 128) =================
    {
        int nt4   = (OUT_PAD / 16) / 4;                               // 2 strips
        int tiles = mtiles * nt4;
        gemm_wmma_f16<HID, OUT_PAD><<<(tiles + GEMM_WAVES - 1) / GEMM_WAVES, 32 * GEMM_WAVES, 0, stream>>>(
            Xh, Wt3, Hf, N, mtiles, nt4);
        int Fq = OUT_PAD / 4;
        agg_init<<<(N * Fq + TB - 1) / TB, TB, 0, stream>>>(Hf, dinv, Agg, N, Fq);
        edge_scatter<<<(E * 32 + TB - 1) / TB, TB, 0, stream>>>(rows, cols, dinv, Hf, Agg, E, Fq, OUT_PAD);
        long long tot = (long long)N * OUT_F;
        bias_out<<<(unsigned)((tot + TB - 1) / TB), TB, 0, stream>>>(Agg, b3, out, N, OUT_F, OUT_PAD);
    }
}
